// FusedHQKAN_30176440221995
// MI455X (gfx1250) — matrix operand order repaired and checked by reading, not compiled, not run on hardware
//
#include <hip/hip_runtime.h>
#include <math.h>

// ---------------------------------------------------------------------------
// Fused QKAN pipeline for MI455X (gfx1250, wave32, WMMA).
//   h   = x @ Wd.T + bd                      (WMMA f16, f32 accum)
//   q   = QKAN(h)  (analytic Bloch rotations collapsed to
//                   z = d1*cp0*cp1 + d2*cp1 + d3*sp0*sp1 + d4*cp0 + d5)
//   out = q @ Wu.T + bu                      (WMMA f16, f32 accum)
// Workspace: f16 Wd / Wu + precomputed per-(i,o) constants (~340 KB).
// ---------------------------------------------------------------------------

typedef __attribute__((ext_vector_type(16))) _Float16 v16h;
typedef __attribute__((ext_vector_type(8)))  _Float16 h8v;
typedef __attribute__((ext_vector_type(8)))  float    v8f;
typedef __attribute__((ext_vector_type(4)))  float    f4v;

#define NROWS 4096
#define DDIM  768
#define WDIM  64

// workspace byte offsets
#define WS_WDH 0                                  // f16 Wd  [64][768]
#define WS_WUH (WDIM * DDIM * 2)                  // f16 Wu  [768][64]
#define WS_QP  (2 * WDIM * DDIM * 2)              // f32 qparams [9][64*64]
#define WS_QB  (WS_QP + 9 * WDIM * WDIM * 4)      // f32 qbias  [64]

// ---------------------------------------------------------------------------
// Prep kernel: f32->f16 weight conversion + QKAN constant folding.
// ---------------------------------------------------------------------------
__global__ __launch_bounds__(256)
void qkan_prep(const float* __restrict__ Wd,
               const float* __restrict__ Wu,
               const float* __restrict__ theta,
               const float* __restrict__ w_enc,
               const float* __restrict__ b_enc,
               const float* __restrict__ scale_base,
               const float* __restrict__ scale_q,
               char* __restrict__ ws)
{
    _Float16* Wdh = (_Float16*)(ws + WS_WDH);
    _Float16* Wuh = (_Float16*)(ws + WS_WUH);
    float*    qp  = (float*)(ws + WS_QP);
    float*    qb  = (float*)(ws + WS_QB);

    const int idx = blockIdx.x * blockDim.x + threadIdx.x;

    if (idx < WDIM * DDIM) {
        Wdh[idx] = (_Float16)Wd[idx];
        Wuh[idx] = (_Float16)Wu[idx];
    }

    if (idx < WDIM * WDIM) {
        const float t0 = theta[idx];
        const float t1 = theta[4096 + idx];
        const float tf = theta[8192 + idx];
        const float w0 = w_enc[idx];
        const float w1 = w_enc[4096 + idx];
        const float b0 = b_enc[idx];
        const float b1 = b_enc[4096 + idx];
        const float sq = scale_q[idx];
        const float sb = scale_base[idx];

        const float ct0 = cosf(t0), st0 = sinf(t0);
        const float ct1 = cosf(t1), st1 = sinf(t1);
        const float ctf = cosf(tf), stf = sinf(tf);

        qp[0 * 4096 + idx] = w0;
        qp[1 * 4096 + idx] = b0;
        qp[2 * 4096 + idx] = w1;
        qp[3 * 4096 + idx] = b1;
        qp[4 * 4096 + idx] = -sq * stf * st0 * ct1;   // d1 * cp0*cp1
        qp[5 * 4096 + idx] = -sq * stf * ct0 * st1;   // d2 * cp1
        qp[6 * 4096 + idx] =  sq * stf * st0;         // d3 * sp0*sp1
        qp[7 * 4096 + idx] = -sq * ctf * st0 * st1;   // d4 * cp0
        qp[8 * 4096 + idx] =  sb;                     // base scale
    }

    if (idx < WDIM) {   // qbias[o] = sum_i sq*ctf*ct0*ct1  (the d5 term)
        float s = 0.0f;
        for (int i = 0; i < WDIM; ++i) {
            const int e = i * WDIM + idx;
            s += scale_q[e] * cosf(theta[8192 + e]) * cosf(theta[e]) * cosf(theta[4096 + e]);
        }
        qb[idx] = s;
    }
}

// ---------------------------------------------------------------------------
// Main fused kernel: 256 threads (8 wave32), 32 rows per workgroup.
// ---------------------------------------------------------------------------
__global__ __launch_bounds__(256)
void qkan_main(const float* __restrict__ x,
               const float* __restrict__ bd,
               const float* __restrict__ bu,
               const char*  __restrict__ ws,
               float* __restrict__ out)
{
    const _Float16* Wdh = (const _Float16*)(ws + WS_WDH);
    const _Float16* Wuh = (const _Float16*)(ws + WS_WUH);
    const float*    qp  = (const float*)(ws + WS_QP);
    const float*    qb  = (const float*)(ws + WS_QB);

    __shared__ float    h_s[32][64];      // pre-activation h
    __shared__ float    s_s[32][64];      // silu(h)
    __shared__ _Float16 q_s[32][64];      // q in f16 for stage-3 A fragments

    const int tid  = threadIdx.x;
    const int wave = tid >> 5;
    const int lane = tid & 31;
    const int hi   = lane >> 4;           // lane half (K-split in WMMA layouts)
    const int l15  = lane & 15;
    const int n0   = blockIdx.x * 32;

    // ---------------- Stage 1: h = x @ Wd.T + bd  (WMMA) -------------------
    {
        const int mt = wave >> 2;                     // row tile 0..1
        const int nt = wave & 3;                      // col tile 0..3
        const float*    xrow = x   + (size_t)(n0 + mt * 16 + l15) * DDIM;
        const _Float16* wrow = Wdh + (size_t)(nt * 16 + l15) * DDIM;

        v8f acc = {0.f, 0.f, 0.f, 0.f, 0.f, 0.f, 0.f, 0.f};
        for (int k = 0; k < DDIM; k += 32) {
            // A fragment: lane holds K = k+hi*16+{0..15} of its row (contiguous,
            // matching B).
            f4v a0 = *(const f4v*)(xrow + k + hi * 16);
            f4v a1 = *(const f4v*)(xrow + k + hi * 16 + 4);
            f4v a2 = *(const f4v*)(xrow + k + hi * 16 + 8);
            f4v a3 = *(const f4v*)(xrow + k + hi * 16 + 12);
            v16h A;
            A[0]  = (_Float16)a0.x; A[1]  = (_Float16)a0.y;
            A[2]  = (_Float16)a0.z; A[3]  = (_Float16)a0.w;
            A[4]  = (_Float16)a1.x; A[5]  = (_Float16)a1.y;
            A[6]  = (_Float16)a1.z; A[7]  = (_Float16)a1.w;
            A[8]  = (_Float16)a2.x; A[9]  = (_Float16)a2.y;
            A[10] = (_Float16)a2.z; A[11] = (_Float16)a2.w;
            A[12] = (_Float16)a3.x; A[13] = (_Float16)a3.y;
            A[14] = (_Float16)a3.z; A[15] = (_Float16)a3.w;
            // B fragment: column (= Wd row) l15, K = k+hi*16+{0..15}, contiguous.
            v16h B = *(const v16h*)(wrow + k + hi * 16);
            acc = __builtin_amdgcn_wmma_f32_16x16x32_f16(
                false, A, false, B, (short)0, acc, false, false);
        }

        const float bdv = bd[nt * 16 + l15];
#pragma unroll
        for (int v = 0; v < 8; ++v) {
            const float hv = acc[v] + bdv;            // C layout: row v+8*hi, col l15
            const int r = mt * 16 + v + 8 * hi;
            const int c = nt * 16 + l15;
            h_s[r][c] = hv;
            s_s[r][c] = hv * __frcp_rn(1.0f + __expf(-hv));   // silu
        }
    }
    __syncthreads();

    // ---------------- Stage 2: QKAN elementwise (trans-heavy) --------------
    {
        const int o = tid & 63;                       // output column
        const int g = tid >> 6;                       // row group (8 rows each)
        float qa[8];
#pragma unroll
        for (int v = 0; v < 8; ++v) qa[v] = 0.f;

        for (int i = 0; i < WDIM; ++i) {
            const float* p = qp + i * 64 + o;         // coalesced across lanes
            const float w0 = p[0 * 4096], b0 = p[1 * 4096];
            const float w1 = p[2 * 4096], b1 = p[3 * 4096];
            const float d1 = p[4 * 4096], d2 = p[5 * 4096];
            const float d3 = p[6 * 4096], d4 = p[7 * 4096];
            const float sb = p[8 * 4096];
#pragma unroll
            for (int nn = 0; nn < 8; ++nn) {
                const int   n  = g * 8 + nn;
                const float hh = h_s[n][i];           // LDS broadcast
                const float ss = s_s[n][i];
                const float phi0 = __builtin_fmaf(w0, hh, b0);
                const float phi1 = __builtin_fmaf(w1, hh, b1);
                const float cp0 = __cosf(phi0), sp0 = __sinf(phi0);
                const float cp1 = __cosf(phi1), sp1 = __sinf(phi1);
                float a = qa[nn];
                a = __builtin_fmaf(__builtin_fmaf(d1, cp0, d2), cp1, a);
                a = __builtin_fmaf(d3 * sp0, sp1, a);
                a = __builtin_fmaf(d4, cp0, a);
                a = __builtin_fmaf(sb, ss, a);        // fused base term
                qa[nn] = a;
            }
        }
        const float qbv = qb[o];
#pragma unroll
        for (int nn = 0; nn < 8; ++nn)
            q_s[g * 8 + nn][o] = (_Float16)(qa[nn] + qbv);
    }
    __syncthreads();

    // ---------------- Stage 3: out = q @ Wu.T + bu  (WMMA) -----------------
    {
        for (int id = wave; id < 96; id += 8) {       // 96 16x16 tiles, 12/wave
            const int mt = id & 1;                    // row tile
            const int nt = id >> 1;                   // col tile 0..47
            v8f acc = {0.f, 0.f, 0.f, 0.f, 0.f, 0.f, 0.f, 0.f};
#pragma unroll
            for (int ks = 0; ks < 2; ++ks) {          // K = 64 in two steps
                const int kk = ks * 32;
                const _Float16* ap = &q_s[mt * 16 + l15][kk + hi * 16];
                h8v alo = *(const h8v*)ap;
                h8v ahi = *(const h8v*)(ap + 8);
                v16h A;
#pragma unroll
                for (int e = 0; e < 8; ++e) { A[e] = alo[e]; A[8 + e] = ahi[e]; }
                v16h B = *(const v16h*)(Wuh + (size_t)(nt * 16 + l15) * 64 + kk + hi * 16);
                acc = __builtin_amdgcn_wmma_f32_16x16x32_f16(
                    false, A, false, B, (short)0, acc, false, false);
            }
            const float buv = bu[nt * 16 + l15];
#pragma unroll
            for (int v = 0; v < 8; ++v) {
                const int r = n0 + mt * 16 + v + 8 * hi;
                out[(size_t)r * DDIM + nt * 16 + l15] = acc[v] + buv;
            }
        }
    }
}

// ---------------------------------------------------------------------------
extern "C" void kernel_launch(void* const* d_in, const int* in_sizes, int n_in,
                              void* d_out, int out_size, void* d_ws, size_t ws_size,
                              hipStream_t stream)
{
    (void)in_sizes; (void)n_in; (void)out_size; (void)ws_size;
    const float* x     = (const float*)d_in[0];
    const float* Wd    = (const float*)d_in[1];
    const float* bd    = (const float*)d_in[2];
    const float* theta = (const float*)d_in[3];
    const float* w_enc = (const float*)d_in[4];
    const float* b_enc = (const float*)d_in[5];
    const float* sbm   = (const float*)d_in[6];
    const float* sqm   = (const float*)d_in[7];
    const float* Wu    = (const float*)d_in[8];
    const float* bu    = (const float*)d_in[9];
    float* out = (float*)d_out;
    char*  ws  = (char*)d_ws;

    qkan_prep<<<(WDIM * DDIM + 255) / 256, 256, 0, stream>>>(
        Wd, Wu, theta, w_enc, b_enc, sbm, sqm, ws);
    qkan_main<<<NROWS / 32, 256, 0, stream>>>(x, bd, bu, ws, out);
}
